// DNAClassifier_87857851007411
// MI455X (gfx1250) — matrix operand order repaired and checked by reading, not compiled, not run on hardware
//
#include <hip/hip_runtime.h>
#include <hip/hip_bf16.h>

// ---------------------------------------------------------------------------
// Types / WMMA helpers (CDNA5 gfx1250, wave32)
// ---------------------------------------------------------------------------
typedef __attribute__((ext_vector_type(16))) _Float16 v16h;
typedef __attribute__((ext_vector_type(8)))  _Float16 v8h;
typedef __attribute__((ext_vector_type(8)))  float    v8f;

#define BB 8
#define SS 256
#define EE 256
#define HH 8
#define DH 32
#define NLYR 2

__device__ __forceinline__ v8f zero8() {
  v8f z = {0.f,0.f,0.f,0.f,0.f,0.f,0.f,0.f};
  return z;
}

__device__ __forceinline__ v8f wmma32(v16h a, v16h b, v8f c) {
  return __builtin_amdgcn_wmma_f32_16x16x32_f16(
      /*neg_a=*/false, a, /*neg_b=*/false, b,
      /*c_mod=*/(short)0, c, /*reuse_a=*/false, /*reuse_b=*/false);
}

// A-fragment (16x32 f16, MxK): lane l(0-15)=row M, halves = K[base..base+7] ++
// K[base+16..base+23], base = (lane>=16 ? 8 : 0).  Caller passes p already at
// (row,K0+base); we read [p..p+7] and [p+16..p+23].
__device__ __forceinline__ v16h frag_a(const _Float16* p) {
  v8h lo = *(const v8h*)p;
  v8h hi = *(const v8h*)(p + 16);
  v16h f;
#pragma unroll
  for (int i = 0; i < 8; ++i) { f[i] = lo[i]; f[i + 8] = hi[i]; }
  return f;
}

// B-fragment (32x16 f16, KxN) from Bt[n][k] storage: lane holds column
// n = lane&15, halves = K[base..base+7] ++ K[base+16..base+23], base = (lane>=16?8:0).
__device__ __forceinline__ v16h frag_b(const _Float16* p) {
  v8h lo = *(const v8h*)p;
  v8h hi = *(const v8h*)(p + 16);
  v16h f;
#pragma unroll
  for (int i = 0; i < 8; ++i) { f[i] = lo[i]; f[i + 8] = hi[i]; }
  return f;
}

// ---------------------------------------------------------------------------
// Weight transpose+convert:  dst[n*K + k] = (f16) src[k*N + n]
// ---------------------------------------------------------------------------
__global__ void k_transpose_h(const float* __restrict__ src,
                              _Float16* __restrict__ dst, int K, int N) {
  int idx = blockIdx.x * 256 + threadIdx.x;
  if (idx < K * N) {
    int k = idx / N, n = idx - k * N;
    dst[(size_t)n * K + k] = (_Float16)src[idx];
  }
}

// ---------------------------------------------------------------------------
// Fused per-segment CNN: conv1(4->128,K3)+ReLU, conv2(128->256,K3)+ReLU,
// maxpool over L.  One block per segment s; loops over the 8 batches reusing
// the f16 W2 A-matrix staged in LDS.  conv2 = 3 shifted GEMMs (K = tap*128+cin)
// 283 KB LDS => exactly one block/WGP, so allow the full VGPR budget
// (launch_bounds (256,1)) to keep all 12 A-fragments resident (no spills).
// ---------------------------------------------------------------------------
#define H1LD 136   // halves per h1T row (128 + pad)
#define W2LD 392   // halves per W2 row (384 + pad)

__global__ __launch_bounds__(256, 1)
void k_conv_fused(const float* __restrict__ x,
                  const float* __restrict__ c1w,
                  const float* __restrict__ c1b,
                  const float* __restrict__ c2w,
                  const float* __restrict__ c2b,
                  float* __restrict__ cnn) {
  extern __shared__ char smem[];
  _Float16* h1T = (_Float16*)smem;              // [258][H1LD]
  _Float16* w2a = h1T + 258 * H1LD;             // [256][W2LD]
  float* w1s = (float*)(w2a + 256 * W2LD);      // 1536
  float* b1s = w1s + 1536;                      // 128
  float* b2s = b1s + 128;                       // 256
  float* xs  = b2s + 256;                       // [4][258]

  const int s    = blockIdx.x;
  const int tid  = threadIdx.x;
  const int lane = tid & 31, wv = tid >> 5;
  const int lane15 = lane & 15;
  const int hi16   = (lane >> 4) & 1;

  // -- stage per-segment weights --------------------------------------------
  for (int i = tid; i < 1536; i += 256) w1s[i] = c1w[s * 1536 + i];
  for (int i = tid; i < 128;  i += 256) b1s[i] = c1b[s * 128 + i];
  for (int i = tid; i < 256;  i += 256) b2s[i] = c2b[s * 256 + i];
  const float* w2seg = c2w + (size_t)s * 256 * 384;   // [cout][cin][tap]
  for (int i = tid; i < 256 * 384; i += 256) {
    __builtin_prefetch(w2seg + i + 4096, 0, 0);       // global_prefetch_b8
    int cout = i / 384, kk = i - cout * 384;          // kk = cin*3 + tap
    int tap = kk % 3, cin = kk / 3;
    w2a[cout * W2LD + tap * 128 + cin] = (_Float16)w2seg[i];
  }
  if (tid < 128) {                                    // zero pad rows l=-1,l=256
    h1T[tid] = (_Float16)0.f;
    h1T[257 * H1LD + tid] = (_Float16)0.f;
  }
  __syncthreads();

  for (int b = 0; b < BB; ++b) {
    // -- stage input segment with zero pads ---------------------------------
    for (int i = tid; i < 4 * 258; i += 256) {
      int ci = i / 258, p = i - ci * 258;
      float v = (p == 0 || p == 257)
                    ? 0.f
                    : x[(((size_t)b * SS + s) * 4 + ci) * 256 + (p - 1)];
      xs[i] = v;
    }
    __syncthreads();

    // -- conv1 + ReLU -> h1T[l+1][co] (f16) ---------------------------------
    for (int i = tid; i < 128 * 256; i += 256) {
      int co = i >> 8, l = i & 255;
      float a = b1s[co];
#pragma unroll
      for (int ci = 0; ci < 4; ++ci)
#pragma unroll
        for (int tap = 0; tap < 3; ++tap)
          a += w1s[(co * 4 + ci) * 3 + tap] * xs[ci * 258 + l + tap];
      h1T[(l + 1) * H1LD + co] = (_Float16)fmaxf(a, 0.f);
    }
    __syncthreads();

    // -- conv2 as WMMA GEMM + ReLU + row-max --------------------------------
    for (int mt = wv; mt < 16; mt += 8) {
      const _Float16* arow = w2a + (mt * 16 + lane15) * W2LD;
      v16h afr[12];
#pragma unroll
      for (int tap = 0; tap < 3; ++tap)
#pragma unroll
        for (int kb = 0; kb < 4; ++kb)
          afr[tap * 4 + kb] = frag_a(arow + tap * 128 + kb * 32 + hi16 * 8);
      float rmax[8] = {0.f,0.f,0.f,0.f,0.f,0.f,0.f,0.f};  // ReLU => max>=0
      for (int nt = 0; nt < 16; ++nt) {
        const _Float16* bbase = h1T + (nt * 16 + lane15) * H1LD + hi16 * 8;
        // B-frag pointer for flattened idx = tap*4+kb:
        //   bbase + tap*H1LD + kb*32
        v8f acc = zero8();
        v16h bcur = frag_b(bbase);   // idx 0 (tap0,kb0)
#pragma unroll
        for (int idx = 0; idx < 12; ++idx) {
          v16h bnext;
          if (idx < 11) {            // prefetch next fragment under the WMMA
            int nidx = idx + 1;
            int ntap = nidx >> 2, nkb = nidx & 3;
            bnext = frag_b(bbase + ntap * H1LD + nkb * 32);
          }
          acc = wmma32(afr[idx], bcur, acc);
          if (idx < 11) bcur = bnext;
        }
#pragma unroll
        for (int r = 0; r < 8; ++r) {
          int m = mt * 16 + hi16 * 8 + r;
          float v = fmaxf(acc[r] + b2s[m], 0.f);
          rmax[r] = fmaxf(rmax[r], v);
        }
      }
#pragma unroll
      for (int r = 0; r < 8; ++r) {   // reduce over the 16-lane half-wave
        float v = rmax[r];
#pragma unroll
        for (int off = 1; off < 16; off <<= 1)
          v = fmaxf(v, __shfl_xor(v, off, 32));
        if (lane15 == 0)
          cnn[(((size_t)b * SS + s) * EE) + mt * 16 + hi16 * 8 + r] = v;
      }
    }
    __syncthreads();
  }
}

// ---------------------------------------------------------------------------
// LayerNorm over E=256: one wave per row (wave32).  g/b may be null.
// ---------------------------------------------------------------------------
__global__ void k_layernorm(const float* __restrict__ x,
                            const float* __restrict__ g,
                            const float* __restrict__ bta,
                            float* __restrict__ outF,
                            _Float16* __restrict__ outH) {
  int lane = threadIdx.x & 31, wv = threadIdx.x >> 5;
  int row = blockIdx.x * 8 + wv;
  const float* xr = x + (size_t)row * EE;
  float v[8], s = 0.f;
#pragma unroll
  for (int i = 0; i < 8; ++i) { v[i] = xr[lane + 32 * i]; s += v[i]; }
#pragma unroll
  for (int off = 1; off < 32; off <<= 1) s += __shfl_xor(s, off, 32);
  float mean = s * (1.f / 256.f);
  float q = 0.f;
#pragma unroll
  for (int i = 0; i < 8; ++i) { float d = v[i] - mean; q += d * d; }
#pragma unroll
  for (int off = 1; off < 32; off <<= 1) q += __shfl_xor(q, off, 32);
  float rstd = rsqrtf(q * (1.f / 256.f) + 1e-5f);
#pragma unroll
  for (int i = 0; i < 8; ++i) {
    int c = lane + 32 * i;
    float o = (v[i] - mean) * rstd;
    if (g) o = o * g[c] + bta[c];
    size_t idx = (size_t)row * EE + c;
    if (outF) outF[idx] = o;
    if (outH) outH[idx] = (_Float16)o;
  }
}

// ---------------------------------------------------------------------------
// Gram matrix G[b][i][j] = <feat[b,i], feat[b,j]>   (WMMA, featH both sides)
// ---------------------------------------------------------------------------
__global__ void k_gram(const _Float16* __restrict__ featH,
                       float* __restrict__ G) {
  int lane = threadIdx.x & 31, wv = threadIdx.x >> 5;
  int lane15 = lane & 15, hi16 = (lane >> 4) & 1;
  int b = blockIdx.x >> 4, it = blockIdx.x & 15;
  const _Float16* base = featH + (size_t)b * SS * EE;
  const _Float16* arow = base + (size_t)(it * 16 + lane15) * EE;
  for (int jt = wv; jt < 16; jt += 8) {
    const _Float16* brow = base + (size_t)(jt * 16 + lane15) * EE;
    v8f acc = zero8();
    v16h a0 = frag_a(arow + hi16 * 8);
    v16h b0 = frag_b(brow + hi16 * 8);
#pragma unroll
    for (int k0 = 0; k0 < EE; k0 += 32) {
      v16h a1, b1;
      if (k0 + 32 < EE) {            // prefetch next K-step under the WMMA
        a1 = frag_a(arow + k0 + 32 + hi16 * 8);
        b1 = frag_b(brow + k0 + 32 + hi16 * 8);
      }
      acc = wmma32(a0, b0, acc);
      if (k0 + 32 < EE) { a0 = a1; b0 = b1; }
    }
#pragma unroll
    for (int r = 0; r < 8; ++r) {
      int i = it * 16 + hi16 * 8 + r, j = jt * 16 + lane15;
      G[(((size_t)b * SS + i) * SS) + j] = acc[r];
    }
  }
}

__global__ void k_init_max(unsigned* p) { if (threadIdx.x == 0) *p = 0u; }

__global__ void k_dist(const float* __restrict__ G, float* __restrict__ dist,
                       unsigned* __restrict__ maxbits) {
  size_t idx = (size_t)blockIdx.x * 256 + threadIdx.x;   // B*S*S total
  int b = (int)(idx >> 16);
  int rem = (int)(idx & 65535);
  int i = rem >> 8, j = rem & 255;
  const float* Gb = G + ((size_t)b << 16);
  float d2 = Gb[i * 256 + i] + Gb[j * 256 + j] - 2.f * Gb[i * 256 + j];
  float d = sqrtf(fmaxf(d2, 0.f));
  dist[idx] = d;
  float m = d;
#pragma unroll
  for (int off = 1; off < 32; off <<= 1) m = fmaxf(m, __shfl_xor(m, off, 32));
  if ((threadIdx.x & 31) == 0) atomicMax(maxbits, __float_as_uint(m));
}

// ---------------------------------------------------------------------------
// Generic WMMA GEMM:  C = A[MxK](f16) * Bt[NxK](f16)^T + bias, opt relu,
// opt f32 residual add, outputs f32 / f16 / f16-transposed-per-batch.
// Block = 128 threads (4 waves), grid = (M/16, N/64); wave -> one 16x16 tile.
// ---------------------------------------------------------------------------
__global__ void k_gemm(const _Float16* __restrict__ A,
                       const _Float16* __restrict__ Bt,
                       const float* __restrict__ bias,
                       const float* __restrict__ resid,
                       float* __restrict__ outF,
                       _Float16* __restrict__ outH,
                       _Float16* __restrict__ outHT,
                       int N, int K, int relu) {
  int lane = threadIdx.x & 31, wv = threadIdx.x >> 5;
  int lane15 = lane & 15, hi16 = (lane >> 4) & 1;
  int mt = blockIdx.x;
  int nt = blockIdx.y * 4 + wv;
  const _Float16* arow = A  + (size_t)(mt * 16 + lane15) * K;
  const _Float16* brow = Bt + (size_t)(nt * 16 + lane15) * K;
  v8f acc = zero8();
  v16h a0 = frag_a(arow + hi16 * 8);
  v16h b0 = frag_b(brow + hi16 * 8);
  for (int k0 = 0; k0 < K; k0 += 32) {
    v16h a1, b1;
    if (k0 + 32 < K) {               // prefetch next K-step under the WMMA
      a1 = frag_a(arow + k0 + 32 + hi16 * 8);
      b1 = frag_b(brow + k0 + 32 + hi16 * 8);
    }
    acc = wmma32(a0, b0, acc);
    if (k0 + 32 < K) { a0 = a1; b0 = b1; }
  }
  int n = nt * 16 + lane15;
  float bv = bias ? bias[n] : 0.f;
#pragma unroll
  for (int r = 0; r < 8; ++r) {
    int m = mt * 16 + hi16 * 8 + r;
    float v = acc[r] + bv;
    if (relu) v = fmaxf(v, 0.f);
    size_t idx = (size_t)m * N + n;
    if (resid) v += resid[idx];
    if (outF) outF[idx] = v;
    if (outH) outH[idx] = (_Float16)v;
    if (outHT) {  // [b][n][s] layout (S=256 rows per batch) for V
      int bb2 = m >> 8, srow = m & 255;
      outHT[((size_t)bb2 * N + n) * SS + srow] = (_Float16)v;
    }
  }
}

// ---------------------------------------------------------------------------
// Attention: block = (b,h) x 64-row i-block.  scores via WMMA (K=DH=32),
// bias affine in dist/rdist, softmax in LDS, then P@V via WMMA.
// ---------------------------------------------------------------------------
#define SCLD 264   // f32 stride for scores rows
#define PBLD 272   // f16 stride for prob rows

__global__ __launch_bounds__(256, 1)
void k_attn(const _Float16* __restrict__ Q,
            const _Float16* __restrict__ Km,
            const _Float16* __restrict__ Vt,
            const float* __restrict__ dist,
            const unsigned* __restrict__ maxbits,
            const float* __restrict__ bias_w,
            const float* __restrict__ bias_b,
            _Float16* __restrict__ attnH) {
  extern __shared__ char smem[];
  float*    sc = (float*)smem;                 // [64][SCLD]
  _Float16* pb = (_Float16*)(sc + 64 * SCLD);  // [64][PBLD]

  int lane = threadIdx.x & 31, wv = threadIdx.x >> 5;
  int lane15 = lane & 15, hi16 = (lane >> 4) & 1;
  int b = blockIdx.x >> 3, h = blockIdx.x & 7;
  int i0 = blockIdx.y * 64;

  float invmax = 1.f / (__uint_as_float(*maxbits) + 1e-6f);
  float w0 = bias_w[0 * HH + h], w1 = bias_w[1 * HH + h];
  float w2 = bias_w[2 * HH + h], w3 = bias_w[3 * HH + h];
  float bbv = bias_b[h];
  const float scale = 0.17677669529663687f;  // 1/sqrt(32)

  const _Float16* Qb = Q  + (size_t)b * SS * EE;
  const _Float16* Kb = Km + (size_t)b * SS * EE;

  // scores: 4 itiles x 16 jtiles, one 16x16x32 WMMA each
  for (int t = wv; t < 64; t += 8) {
    int it = t >> 4, jt = t & 15;
    const _Float16* arow = Qb + (size_t)(i0 + it * 16 + lane15) * EE + h * DH;
    const _Float16* brow = Kb + (size_t)(jt * 16 + lane15) * EE + h * DH;
    v8f acc = wmma32(frag_a(arow + hi16 * 8), frag_b(brow + hi16 * 8), zero8());
    int j = jt * 16 + lane15;
#pragma unroll
    for (int r = 0; r < 8; ++r) {
      int mi = it * 16 + hi16 * 8 + r;
      int ia = i0 + mi;
      float dn = dist[(((size_t)b * SS + ia) * SS) + j] * invmax;
      float rd = fabsf((float)(ia - j)) * (1.f / 255.f);
      float bv = w0 * dn + w1 * rd + w2 * (1.f - rd) + w3 * (1.f - dn) + bbv;
      sc[mi * SCLD + j] = acc[r] * scale + bv;
    }
  }
  __syncthreads();

  // softmax: each wave handles 8 rows
  for (int rr = 0; rr < 8; ++rr) {
    int row = wv * 8 + rr;
    float vv[8], m = -1e30f;
#pragma unroll
    for (int c = 0; c < 8; ++c) {
      vv[c] = sc[row * SCLD + lane + 32 * c];
      m = fmaxf(m, vv[c]);
    }
#pragma unroll
    for (int off = 1; off < 32; off <<= 1) m = fmaxf(m, __shfl_xor(m, off, 32));
    float s = 0.f;
#pragma unroll
    for (int c = 0; c < 8; ++c) { vv[c] = __expf(vv[c] - m); s += vv[c]; }
#pragma unroll
    for (int off = 1; off < 32; off <<= 1) s += __shfl_xor(s, off, 32);
    float inv = 1.f / s;
#pragma unroll
    for (int c = 0; c < 8; ++c)
      pb[row * PBLD + lane + 32 * c] = (_Float16)(vv[c] * inv);
  }
  __syncthreads();

  // P @ V : 4 itiles x 2 dtiles = 8 tiles, one per wave; K-loop = 256/32
  {
    int it = wv >> 1, dt = wv & 1;
    const _Float16* arow = pb + (it * 16 + lane15) * PBLD;
    const _Float16* brow =
        Vt + ((size_t)b * EE + h * DH + dt * 16 + lane15) * SS;
    v8f acc = zero8();
    v16h a0 = frag_a(arow + hi16 * 8);
    v16h b0 = frag_b(brow + hi16 * 8);
#pragma unroll
    for (int k0 = 0; k0 < SS; k0 += 32) {
      v16h a1, b1;
      if (k0 + 32 < SS) {
        a1 = frag_a(arow + k0 + 32 + hi16 * 8);
        b1 = frag_b(brow + k0 + 32 + hi16 * 8);
      }
      acc = wmma32(a0, b0, acc);
      if (k0 + 32 < SS) { a0 = a1; b0 = b1; }
    }
#pragma unroll
    for (int r = 0; r < 8; ++r) {
      int mi = it * 16 + hi16 * 8 + r;
      attnH[(((size_t)b * SS + i0 + mi) * EE) + h * DH + dt * 16 + lane15] =
          (_Float16)acc[r];
    }
  }
}

// ---------------------------------------------------------------------------
// Final: per-batch chunked max-pool (8 chunks of 32 segments) + FC to 2 logits
// ---------------------------------------------------------------------------
__global__ void k_final(const float* __restrict__ hbuf,
                        const float* __restrict__ fcw,
                        const float* __restrict__ fcb,
                        float* __restrict__ out) {
  __shared__ float pooled[2048];
  __shared__ float red[256][2];
  int b = blockIdx.x, tid = threadIdx.x;
  for (int i = tid; i < 2048; i += 256) {
    int ns = i >> 8, e = i & 255;
    float m = -1e30f;
    for (int k = 0; k < 32; ++k)
      m = fmaxf(m, hbuf[(((size_t)b * SS + ns * 32 + k) * EE) + e]);
    pooled[i] = m;
  }
  __syncthreads();
  float s0 = 0.f, s1 = 0.f;
  for (int i = tid; i < 2048; i += 256) {
    float p = pooled[i];
    s0 += p * fcw[i * 2 + 0];
    s1 += p * fcw[i * 2 + 1];
  }
  red[tid][0] = s0; red[tid][1] = s1;
  __syncthreads();
  for (int st = 128; st > 0; st >>= 1) {
    if (tid < st) { red[tid][0] += red[tid + st][0]; red[tid][1] += red[tid + st][1]; }
    __syncthreads();
  }
  if (tid == 0) {
    out[b * 2 + 0] = red[0][0] + fcb[0];
    out[b * 2 + 1] = red[0][1] + fcb[1];
  }
}

// ---------------------------------------------------------------------------
// Host orchestration
// ---------------------------------------------------------------------------
extern "C" void kernel_launch(void* const* d_in, const int* in_sizes, int n_in,
                              void* d_out, int out_size, void* d_ws, size_t ws_size,
                              hipStream_t stream) {
  (void)in_sizes; (void)n_in; (void)out_size; (void)ws_size;
  const float* x    = (const float*)d_in[0];
  const float* c1w  = (const float*)d_in[1];
  const float* c1b  = (const float*)d_in[2];
  const float* c2w  = (const float*)d_in[3];
  const float* c2b  = (const float*)d_in[4];
  const float* ln1g = (const float*)d_in[5];
  const float* ln1b = (const float*)d_in[6];
  const float* wq   = (const float*)d_in[7];
  const float* bq   = (const float*)d_in[8];
  const float* wk   = (const float*)d_in[9];
  const float* bk   = (const float*)d_in[10];
  const float* wv   = (const float*)d_in[11];
  const float* bv   = (const float*)d_in[12];
  const float* wo   = (const float*)d_in[13];
  const float* bo   = (const float*)d_in[14];
  const float* bw   = (const float*)d_in[15];
  const float* bbb  = (const float*)d_in[16];
  const float* ln2g = (const float*)d_in[17];
  const float* ln2b = (const float*)d_in[18];
  const float* fw1  = (const float*)d_in[19];
  const float* fb1  = (const float*)d_in[20];
  const float* fw2  = (const float*)d_in[21];
  const float* fb2  = (const float*)d_in[22];
  const float* fcw  = (const float*)d_in[23];
  const float* fcb  = (const float*)d_in[24];
  float* out = (float*)d_out;

  const int M = BB * SS;  // 2048 token rows
  char* p = (char*)d_ws;
  auto alloc = [&](size_t bytes) {
    char* r = p;
    p += (bytes + 255) & ~(size_t)255;
    return r;
  };
  float*    cnn   = (float*)alloc((size_t)M * EE * 4);
  float*    hbuf  = (float*)alloc((size_t)M * EE * 4);
  _Float16* featH = (_Float16*)alloc((size_t)M * EE * 2);
  float*    G     = (float*)alloc((size_t)BB * SS * SS * 4);
  float*    distB = (float*)alloc((size_t)BB * SS * SS * 4);
  unsigned* maxb  = (unsigned*)alloc(256);
  _Float16* wtQ   = (_Float16*)alloc((size_t)NLYR * EE * EE * 2);
  _Float16* wtK   = (_Float16*)alloc((size_t)NLYR * EE * EE * 2);
  _Float16* wtV   = (_Float16*)alloc((size_t)NLYR * EE * EE * 2);
  _Float16* wtO   = (_Float16*)alloc((size_t)NLYR * EE * EE * 2);
  _Float16* wt1   = (_Float16*)alloc((size_t)NLYR * EE * 512 * 2);
  _Float16* wt2   = (_Float16*)alloc((size_t)NLYR * 512 * EE * 2);
  _Float16* hnH   = (_Float16*)alloc((size_t)M * EE * 2);
  _Float16* qH    = (_Float16*)alloc((size_t)M * EE * 2);
  _Float16* kH    = (_Float16*)alloc((size_t)M * EE * 2);
  _Float16* vtH   = (_Float16*)alloc((size_t)M * EE * 2);
  _Float16* aH    = (_Float16*)alloc((size_t)M * EE * 2);
  _Float16* ffnH  = (_Float16*)alloc((size_t)M * 512 * 2);

  // 1) weight prep (transpose + f16 convert)
  for (int l = 0; l < NLYR; ++l) {
    k_transpose_h<<<256, 256, 0, stream>>>(wq + l * EE * EE, wtQ + l * EE * EE, EE, EE);
    k_transpose_h<<<256, 256, 0, stream>>>(wk + l * EE * EE, wtK + l * EE * EE, EE, EE);
    k_transpose_h<<<256, 256, 0, stream>>>(wv + l * EE * EE, wtV + l * EE * EE, EE, EE);
    k_transpose_h<<<256, 256, 0, stream>>>(wo + l * EE * EE, wtO + l * EE * EE, EE, EE);
    k_transpose_h<<<512, 256, 0, stream>>>(fw1 + l * EE * 512, wt1 + l * EE * 512, EE, 512);
    k_transpose_h<<<512, 256, 0, stream>>>(fw2 + l * 512 * EE, wt2 + l * 512 * EE, 512, EE);
  }

  // 2) fused per-segment CNN (WMMA conv2 in 283 KB LDS)
  size_t convLds = (size_t)(258 * H1LD + 256 * W2LD) * 2 +
                   (size_t)(1536 + 128 + 256 + 4 * 258) * 4;
  k_conv_fused<<<SS, 256, convLds, stream>>>(x, c1w, c1b, c2w, c2b, cnn);

  // 3) feature layernorm (no affine)
  k_layernorm<<<M / 8, 256, 0, stream>>>(cnn, nullptr, nullptr, hbuf, featH);

  // 4) distance bias: Gram via WMMA, then dist + global max
  k_gram<<<BB * 16, 256, 0, stream>>>(featH, G);
  k_init_max<<<1, 32, 0, stream>>>(maxb);
  k_dist<<<(BB * SS * SS) / 256, 256, 0, stream>>>(G, distB, maxb);

  // 5) transformer layers
  dim3 g256(M / 16, EE / 64);   // N = 256
  dim3 g512(M / 16, 512 / 64);  // N = 512
  size_t attnLds = (size_t)64 * SCLD * 4 + (size_t)64 * PBLD * 2;
  for (int l = 0; l < NLYR; ++l) {
    k_layernorm<<<M / 8, 256, 0, stream>>>(hbuf, ln1g + l * EE, ln1b + l * EE,
                                           nullptr, hnH);
    k_gemm<<<g256, 128, 0, stream>>>(hnH, wtQ + l * EE * EE, bq + l * EE,
                                     nullptr, nullptr, qH, nullptr, EE, EE, 0);
    k_gemm<<<g256, 128, 0, stream>>>(hnH, wtK + l * EE * EE, bk + l * EE,
                                     nullptr, nullptr, kH, nullptr, EE, EE, 0);
    k_gemm<<<g256, 128, 0, stream>>>(hnH, wtV + l * EE * EE, bv + l * EE,
                                     nullptr, nullptr, nullptr, vtH, EE, EE, 0);
    k_attn<<<dim3(BB * HH, SS / 64), 256, attnLds, stream>>>(
        qH, kH, vtH, distB, maxb, bw + l * 4 * HH, bbb + l * HH, aH);
    k_gemm<<<g256, 128, 0, stream>>>(aH, wtO + l * EE * EE, bo + l * EE,
                                     hbuf, hbuf, nullptr, nullptr, EE, EE, 0);
    k_layernorm<<<M / 8, 256, 0, stream>>>(hbuf, ln2g + l * EE, ln2b + l * EE,
                                           nullptr, hnH);
    k_gemm<<<g512, 128, 0, stream>>>(hnH, wt1 + l * EE * 512, fb1 + l * 512,
                                     nullptr, nullptr, ffnH, nullptr, 512, EE, 1);
    k_gemm<<<g256, 128, 0, stream>>>(ffnH, wt2 + l * 512 * EE, fb2 + l * EE,
                                     hbuf, hbuf, nullptr, nullptr, EE, 512, 0);
  }

  // 6) chunked max-pool + FC
  k_final<<<BB, 256, 0, stream>>>(hbuf, fcw, fcb, out);
}
